// HeteroGAT_85744727097415
// MI455X (gfx1250) — matrix operand order repaired and checked by reading, not compiled, not run on hardware
//
#include <hip/hip_runtime.h>
#include <math.h>

typedef __attribute__((ext_vector_type(16))) _Float16 v16h;
typedef __attribute__((ext_vector_type(8)))  _Float16 v8h;
typedef __attribute__((ext_vector_type(8)))  float    v8f;

#define CAT16(x, y) __builtin_shufflevector((x), (y), 0, 1, 2, 3, 4, 5, 6, 7, 8, 9, 10, 11, 12, 13, 14, 15)

__device__ __forceinline__ float f_or0(float v) { return v == v ? v : 0.0f; }

static inline int nblk(long n, int b) { return (int)((n + b - 1) / b); }

// ---------------------------------------------------------------- fills / preps
__global__ void k_fill(float* __restrict__ p, float v, long n) {
  long t = (long)blockIdx.x * blockDim.x + threadIdx.x;
  if (t < n) p[t] = v;
}

// f32 -> f16 copy with nan_to_num
__global__ void k_prep_h(const float* __restrict__ in, _Float16* __restrict__ out, long n) {
  long t = (long)blockIdx.x * blockDim.x + threadIdx.x;
  if (t < n) out[t] = (_Float16)f_or0(in[t]);
}

// BT[n][k] = (f16) B[k][n]   (B row-major [K,N])
__global__ void k_prep_bt(const float* __restrict__ B, _Float16* __restrict__ BT,
                          int K, int N, long total) {
  long t = (long)blockIdx.x * blockDim.x + threadIdx.x;
  if (t >= total) return;
  int n = (int)(t / K), k = (int)(t % K);
  BT[t] = (_Float16)B[(long)k * N + n];
}

// CT[col][k] = (f16) CE[k][col] for k<16, 0 for 16<=k<32  (CE is [16,128] f32)
__global__ void k_prep_ct(const float* __restrict__ CE, _Float16* __restrict__ CT) {
  int t = blockIdx.x * blockDim.x + threadIdx.x;
  if (t >= 128 * 32) return;
  int col = t >> 5, k = t & 31;
  CT[t] = (_Float16)(k < 16 ? CE[(long)k * 128 + col] : 0.0f);
}

// out[n][c] = b0[c] (+ b1[c])
__global__ void k_init_bias(float* __restrict__ out, const float* __restrict__ b0,
                            const float* __restrict__ b1, long total, int C) {
  long t = (long)blockIdx.x * blockDim.x + threadIdx.x;
  if (t >= total) return;
  int c = (int)(t % C);
  out[t] = b0[c] + (b1 ? b1[c] : 0.0f);
}

// ---------------------------------------------------------------- WMMA GEMM
// C[M,N] = A[M,K](f16, row stride K) @ BT[N,K](f16 transposed); N%128==0.
// K compile-time: all B addresses = uniform base + 32-bit lane offset + imm,
// fully unrolled -> pure global_load_b128 + v_wmma stream, no VALU in loop.
template <int K>
__global__ void k_gemm_wmma_t(const _Float16* __restrict__ A, const _Float16* __restrict__ BT,
                              float* __restrict__ C, int M, int N) {
  int wave = (int)(((long)blockIdx.x * blockDim.x + threadIdx.x) >> 5);
  int lane = threadIdx.x & 31;
  int ngroups = N >> 7;
  int mtiles = (M + 15) >> 4;
  int mt = wave / ngroups, ng = wave - mt * ngroups;
  if (mt >= mtiles) return;
  int half = lane >> 4, l16 = lane & 15;
  int row = mt * 16 + l16;
  int rowc = row < M ? row : M - 1;  // clamp: junk rows never stored
  int aoff = rowc * K + half * 8;
  int boff = (ng * 128 + l16) * K + half * 8;
  v8f zero = {};
  v8f acc[8];
#pragma unroll
  for (int t = 0; t < 8; ++t) acc[t] = zero;
#pragma unroll
  for (int k0 = 0; k0 < K; k0 += 32) {
    v16h a = CAT16(*(const v8h*)(A + aoff + k0),
                   *(const v8h*)(A + aoff + k0 + 16));
#pragma unroll
    for (int t = 0; t < 8; ++t) {
      v16h b = CAT16(*(const v8h*)(BT + boff + t * 16 * K + k0),
                     *(const v8h*)(BT + boff + t * 16 * K + k0 + 16));
      acc[t] = __builtin_amdgcn_wmma_f32_16x16x32_f16(false, a, false, b, (short)0,
                                                      acc[t], false, false);
    }
  }
#pragma unroll
  for (int r = 0; r < 8; ++r) {
    int m = mt * 16 + r + half * 8;
    if (m < M) {
      float* crow = C + (long)m * N + ng * 128 + l16;
#pragma unroll
      for (int t = 0; t < 8; ++t) crow[t * 16] = acc[t][r];
    }
  }
}

// ---------------------------------------------------------------- attention dot
__global__ void k_attdot(const float* __restrict__ H, const float* __restrict__ att,
                         float* __restrict__ out, int N) {
  int wave = (int)(((long)blockIdx.x * blockDim.x + threadIdx.x) >> 5);
  int lane = threadIdx.x & 31;
  int n = wave >> 1, h = wave & 1;
  if (n >= N) return;
  const float* row = H + (long)n * 256 + h * 128;
  const float* a = att + h * 128;
  float s = 0.0f;
#pragma unroll
  for (int i = 0; i < 4; ++i) s += row[lane + 32 * i] * a[lane + 32 * i];
#pragma unroll
  for (int off = 16; off; off >>= 1) s += __shfl_down(s, off, 32);
  if (lane == 0) out[n * 2 + h] = s;
}

// ordered-int float atomic max (locations pre-initialized to -inf)
__device__ __forceinline__ void atomicMaxF(float* addr, float v) {
  int iv = __float_as_int(v);
  if (iv >= 0) atomicMax((int*)addr, iv);
  else         atomicMin((unsigned int*)addr, (unsigned int)iv);
}

__global__ void k_edge_max(const int* __restrict__ ei, int E,
                           const float* __restrict__ as, const float* __restrict__ ad,
                           float* __restrict__ emax) {
  int t = blockIdx.x * blockDim.x + threadIdx.x;
  if (t >= E * 2) return;
  int e = t >> 1, h = t & 1;
  int s = ei[e], d = ei[E + e];
  float v = as[s * 2 + h] + ad[d * 2 + h];
  v = v > 0.0f ? v : 0.2f * v;
  atomicMaxF(&emax[d * 2 + h], v);
}

__global__ void k_emax_fix(float* __restrict__ emax, int n) {
  int t = blockIdx.x * blockDim.x + threadIdx.x;
  if (t >= n) return;
  float v = emax[t];
  if (!(v > -1.0e30f)) emax[t] = 0.0f;
}

__global__ void k_edge_pe(const int* __restrict__ ei, int E,
                          const float* __restrict__ as, const float* __restrict__ ad,
                          const float* __restrict__ emax, float* __restrict__ pe,
                          float* __restrict__ den) {
  int t = blockIdx.x * blockDim.x + threadIdx.x;
  if (t >= E * 2) return;
  int e = t >> 1, h = t & 1;
  int s = ei[e], d = ei[E + e];
  float v = as[s * 2 + h] + ad[d * 2 + h];
  v = v > 0.0f ? v : 0.2f * v;
  float p = __expf(v - emax[d * 2 + h]);
  pe[t] = p;
  atomicAdd(&den[d * 2 + h], p);
}

__global__ void k_edge_acc(const int* __restrict__ ei, int E,
                           const float* __restrict__ pe, const float* __restrict__ den,
                           const float* __restrict__ hs, float* __restrict__ out) {
  long t = (long)blockIdx.x * blockDim.x + threadIdx.x;
  if (t >= (long)E * 64) return;
  int e = (int)(t >> 6);
  int c0 = (int)(t & 63) * 4;
  int s = ei[e], d = ei[E + e];
  int h = c0 >> 7;
  float alpha = pe[e * 2 + h] / (den[d * 2 + h] + 1e-16f);
  float4 hv = *(const float4*)(hs + (long)s * 256 + c0);
  float* o = out + (long)d * 256 + c0;
  atomicAdd(o + 0, hv.x * alpha);
  atomicAdd(o + 1, hv.y * alpha);
  atomicAdd(o + 2, hv.z * alpha);
  atomicAdd(o + 3, hv.w * alpha);
}

__global__ void k_combine(float* __restrict__ acc, const float* __restrict__ res,
                          float scale, long total) {
  long t = (long)blockIdx.x * blockDim.x + threadIdx.x;
  if (t >= total) return;
  float v = acc[t] * scale;
  if (res) v += res[t];
  acc[t] = v;
}

__global__ void k_colstats(const float* __restrict__ X, int N, int C,
                           float* __restrict__ sum, float* __restrict__ sumsq) {
  int c = threadIdx.x;
  float s = 0.0f, ss = 0.0f;
  for (int n = blockIdx.x; n < N; n += gridDim.x) {
    float v = X[(long)n * C + c];
    s += v; ss += v * v;
  }
  atomicAdd(&sum[c], s);
  atomicAdd(&sumsq[c], ss);
}

__global__ void k_bn_final(const float* __restrict__ sum, const float* __restrict__ sumsq,
                           int N, int C, const float* __restrict__ g,
                           const float* __restrict__ b, const float* __restrict__ dbias,
                           float* __restrict__ scale, float* __restrict__ shift) {
  int c = blockIdx.x * blockDim.x + threadIdx.x;
  if (c >= C) return;
  float d = dbias ? dbias[c] : 0.0f;
  float m1 = sum[c] / N;
  float mean = m1 + d;
  float ex2 = sumsq[c] / N + 2.0f * d * m1 + d * d;
  float var = ex2 - mean * mean;
  float inv = rsqrtf(var + 1e-5f);
  scale[c] = inv * g[c];
  shift[c] = b[c] - mean * inv * g[c];
}

__global__ void k_bn_apply(float* __restrict__ X, const float* __restrict__ scale,
                           const float* __restrict__ shift, long total, int C,
                           float slope) {
  long t = (long)blockIdx.x * blockDim.x + threadIdx.x;
  if (t >= total) return;
  int c = (int)(t % C);
  float v = X[t] * scale[c] + shift[c];
  X[t] = v > 0.0f ? v : slope * v;
}

__global__ void k_edge_dvec(const float* __restrict__ epb, const float* __restrict__ W1mid,
                            const float* __restrict__ b1, float* __restrict__ dv) {
  int c = threadIdx.x;
  float s = b1[c];
  for (int j = 0; j < 128; ++j) s += epb[j] * W1mid[(long)j * 128 + c];
  dv[c] = s;
}

// ---------------------------------------------------------------- fused edge head
// One wave handles 16 edges x all 128 output cols; gathered A fragment loaded
// once per K-step, 8 independent accumulators; all addresses are uniform base
// + 32-bit offset + compile-time immediate (fully unrolled).
// mode 0: BN column stats (raw sums; biases folded analytically later)
// mode 1: out[e] = lrelu((h+dv)*scale+shift) . W2 + b2
__global__ void k_edge_head(const int* __restrict__ ei, int E,
                            const _Float16* __restrict__ XS, const _Float16* __restrict__ XD,
                            const _Float16* __restrict__ EAH,
                            const _Float16* __restrict__ WT1s, const _Float16* __restrict__ WT1d,
                            const _Float16* __restrict__ CT, const float* __restrict__ dv,
                            int mode, float* __restrict__ sum, float* __restrict__ sumsq,
                            const float* __restrict__ scale, const float* __restrict__ shift,
                            const float* __restrict__ W2, const float* __restrict__ b2p,
                            float* __restrict__ out) {
  int wave = (int)(((long)blockIdx.x * blockDim.x + threadIdx.x) >> 5);
  int lane = threadIdx.x & 31;
  int e0 = wave * 16;
  if (e0 >= E) return;
  int half = lane >> 4, l16 = lane & 15;
  int erow = e0 + l16;
  int erc = erow < E ? erow : E - 1;  // clamp: junk rows masked at epilogue
  int src = ei[erc];
  int dst = ei[E + erc];
  int xsoff = src * 256 + half * 8;    // 32-bit offsets off uniform bases
  int xdoff = dst * 256 + half * 8;
  int eaoff = erc * 16 + half * 8;
  int wb = l16 * 256 + half * 8;
  int cb = l16 * 32 + half * 8;
  v8f zero = {};
  v8f acc[8];
#pragma unroll
  for (int t = 0; t < 8; ++t) acc[t] = zero;

  // x_s block (K=256, W1 rows 0..255)
#pragma unroll
  for (int k0 = 0; k0 < 256; k0 += 32) {
    v16h a = CAT16(*(const v8h*)(XS + xsoff + k0),
                   *(const v8h*)(XS + xsoff + k0 + 16));
#pragma unroll
    for (int t = 0; t < 8; ++t) {
      v16h b = CAT16(*(const v8h*)(WT1s + wb + t * 4096 + k0),
                     *(const v8h*)(WT1s + wb + t * 4096 + k0 + 16));
      acc[t] = __builtin_amdgcn_wmma_f32_16x16x32_f16(false, a, false, b, (short)0,
                                                      acc[t], false, false);
    }
  }
  // x_d block (K=256, W1 rows 384..639)
#pragma unroll
  for (int k0 = 0; k0 < 256; k0 += 32) {
    v16h a = CAT16(*(const v8h*)(XD + xdoff + k0),
                   *(const v8h*)(XD + xdoff + k0 + 16));
#pragma unroll
    for (int t = 0; t < 8; ++t) {
      v16h b = CAT16(*(const v8h*)(WT1d + wb + t * 4096 + k0),
                     *(const v8h*)(WT1d + wb + t * 4096 + k0 + 16));
      acc[t] = __builtin_amdgcn_wmma_f32_16x16x32_f16(false, a, false, b, (short)0,
                                                      acc[t], false, false);
    }
  }
  // edge-attr block: K padded 16->32 (CT pre-padded with zeros)
  {
    v8h z = {};
    v16h a = CAT16(*(const v8h*)(EAH + eaoff), z);
#pragma unroll
    for (int t = 0; t < 8; ++t) {
      v16h b = CAT16(*(const v8h*)(CT + cb + t * 512),
                     *(const v8h*)(CT + cb + t * 512 + 16));
      acc[t] = __builtin_amdgcn_wmma_f32_16x16x32_f16(false, a, false, b, (short)0,
                                                      acc[t], false, false);
    }
  }

  if (mode == 0) {
#pragma unroll
    for (int t = 0; t < 8; ++t) {
      int col = t * 16 + l16;
      float s = 0.0f, ss = 0.0f;
#pragma unroll
      for (int r = 0; r < 8; ++r) {
        int m = r + half * 8;
        float v = (e0 + m) < E ? acc[t][r] : 0.0f;
        s += v; ss += v * v;
      }
      s += __shfl_xor(s, 16, 32);
      ss += __shfl_xor(ss, 16, 32);
      if (half == 0) { atomicAdd(&sum[col], s); atomicAdd(&sumsq[col], ss); }
    }
  } else {
    float rowacc[8];
#pragma unroll
    for (int r = 0; r < 8; ++r) rowacc[r] = 0.0f;
#pragma unroll
    for (int t = 0; t < 8; ++t) {
      int col = t * 16 + l16;
      float dvc = dv[col], sc = scale[col], sh = shift[col], w2 = W2[col];
#pragma unroll
      for (int r = 0; r < 8; ++r) {
        float v = (acc[t][r] + dvc) * sc + sh;
        v = v > 0.0f ? v : 0.01f * v;
        rowacc[r] += v * w2;
      }
    }
    float b2 = b2p[0];
#pragma unroll
    for (int r = 0; r < 8; ++r) {
      float v = rowacc[r];
#pragma unroll
      for (int off = 1; off < 16; off <<= 1) v += __shfl_xor(v, off, 32);
      int e = e0 + r + half * 8;
      if (l16 == 0 && e < E) out[e] = v + b2;
    }
  }
}

// ---------------------------------------------------------------- host side
struct GatP { const float *Wsrc, *Wdst, *asrc, *adst, *bias; };

static void gemm(hipStream_t st, const _Float16* A, const float* Bf,
                 _Float16* BT, float* C, int M, int N, int K) {
  k_prep_bt<<<nblk((long)N * K, 256), 256, 0, st>>>(Bf, BT, K, N, (long)N * K);
  long waves = (long)((M + 15) / 16) * (N / 128);
  int blocks = nblk(waves * 32, 256);
  switch (K) {
    case 32:  k_gemm_wmma_t<32><<<blocks, 256, 0, st>>>(A, BT, C, M, N); break;
    case 64:  k_gemm_wmma_t<64><<<blocks, 256, 0, st>>>(A, BT, C, M, N); break;
    case 128: k_gemm_wmma_t<128><<<blocks, 256, 0, st>>>(A, BT, C, M, N); break;
    default:  k_gemm_wmma_t<256><<<blocks, 256, 0, st>>>(A, BT, C, M, N); break;
  }
}

static void run_rel(hipStream_t st, const _Float16* Xs, int Ks, int Ns,
                    const _Float16* Xd, int Kd, int Nd, const int* ei, int E,
                    const GatP& g, float* accum, _Float16* BT, float* HS, float* HD,
                    float* AS, float* AD, float* EMAX, float* DEN, float* PE) {
  gemm(st, Xs, g.Wsrc, BT, HS, Ns, 256, Ks);
  gemm(st, Xd, g.Wdst, BT, HD, Nd, 256, Kd);
  k_attdot<<<nblk((long)Ns * 2 * 32, 256), 256, 0, st>>>(HS, g.asrc, AS, Ns);
  k_attdot<<<nblk((long)Nd * 2 * 32, 256), 256, 0, st>>>(HD, g.adst, AD, Nd);
  k_fill<<<nblk(Nd * 2, 256), 256, 0, st>>>(EMAX, -INFINITY, Nd * 2);
  k_fill<<<nblk(Nd * 2, 256), 256, 0, st>>>(DEN, 0.0f, Nd * 2);
  k_edge_max<<<nblk((long)E * 2, 256), 256, 0, st>>>(ei, E, AS, AD, EMAX);
  k_emax_fix<<<nblk(Nd * 2, 256), 256, 0, st>>>(EMAX, Nd * 2);
  k_edge_pe<<<nblk((long)E * 2, 256), 256, 0, st>>>(ei, E, AS, AD, EMAX, PE, DEN);
  k_edge_acc<<<nblk((long)E * 64, 256), 256, 0, st>>>(ei, E, PE, DEN, HS, accum);
}

extern "C" void kernel_launch(void* const* d_in, const int* in_sizes, int n_in,
                              void* d_out, int out_size, void* d_ws, size_t ws_size,
                              hipStream_t stream) {
  (void)n_in; (void)out_size; (void)ws_size;
  const float* xreq = (const float*)d_in[0];
  const float* xveh = (const float*)d_in[1];
  const float* ea[3] = { (const float*)d_in[5], (const float*)d_in[6], (const float*)d_in[7] };
  int NR = in_sizes[0] / 64, NV = in_sizes[1] / 32;
  int E_[3] = { in_sizes[2] / 2, in_sizes[3] / 2, in_sizes[4] / 2 };
  const int* EI[3] = { (const int*)d_in[2], (const int*)d_in[3], (const int*)d_in[4] };

  int p = 8;
  GatP L[2][3];
  const float *bng[2][2], *bnb[2][2];
  for (int l = 0; l < 2; ++l) {
    for (int r = 0; r < 3; ++r) {
      L[l][r].Wsrc = (const float*)d_in[p++];
      L[l][r].Wdst = (const float*)d_in[p++];
      L[l][r].asrc = (const float*)d_in[p++];
      L[l][r].adst = (const float*)d_in[p++];
      L[l][r].bias = (const float*)d_in[p++];
    }
    bng[l][0] = (const float*)d_in[p++]; bnb[l][0] = (const float*)d_in[p++];
    bng[l][1] = (const float*)d_in[p++]; bnb[l][1] = (const float*)d_in[p++];
  }
  const float *epW[3], *epB[3];
  for (int r = 0; r < 3; ++r) { epW[r] = (const float*)d_in[p++]; epB[r] = (const float*)d_in[p++]; }
  const float* lin1W = (const float*)d_in[p++];
  const float* lin1b = (const float*)d_in[p++];
  const float* lin2W = (const float*)d_in[p++];
  const float* lin2b = (const float*)d_in[p++];
  const float* fng = (const float*)d_in[p++];
  const float* fnb = (const float*)d_in[p++];

  // workspace carve-up (256B aligned)
  char* w = (char*)d_ws;
  auto carveF = [&](long nf) { float* r = (float*)w; w += ((nf * 4 + 255) / 256) * 256; return r; };
  auto carveH = [&](long nh) { _Float16* r = (_Float16*)w; w += ((nh * 2 + 255) / 256) * 256; return r; };
  float* XRb[2] = { carveF((long)NR * 256), carveF((long)NR * 256) };
  float* XVb[2] = { carveF((long)NV * 256), carveF((long)NV * 256) };
  float* HS = carveF((long)NR * 256);
  float* HD = carveF((long)NR * 256);
  float* AS = carveF(NR * 2);
  float* AD = carveF(NR * 2);
  float* EMAX = carveF(NR * 2);
  float* DEN = carveF(NR * 2);
  float* PE = carveF((long)E_[0] * 2);
  float* SUM = carveF(512);
  float* SCALE = carveF(256);
  float* SHIFT = carveF(256);
  float* CE = carveF(16 * 128);
  float* DV = carveF(128);
  _Float16* BT = carveH(256 * 256);             // transposed f16 weight scratch
  _Float16* XRH = carveH((long)NR * 256);       // f16 activations
  _Float16* XVH = carveH((long)NV * 256);
  _Float16* X0RH = carveH((long)NR * 64);
  _Float16* X0VH = carveH((long)NV * 32);
  _Float16* WT1s = carveH(128 * 256);
  _Float16* WT1d = carveH(128 * 256);
  _Float16* CT = carveH(128 * 32);
  _Float16* EPWH = carveH(16 * 128);
  _Float16* EAH = carveH((long)E_[0] * 16);

  // f16 copies of raw inputs (folds nan_to_num)
  k_prep_h<<<nblk((long)NR * 64, 256), 256, 0, stream>>>(xreq, X0RH, (long)NR * 64);
  k_prep_h<<<nblk((long)NV * 32, 256), 256, 0, stream>>>(xveh, X0VH, (long)NV * 32);

  const _Float16* xrh = X0RH; int kr = 64;
  const _Float16* xvh = X0VH; int kv = 32;
  const float* xr_res = nullptr;
  const float* xv_res = nullptr;
  for (int l = 0; l < 2; ++l) {
    float* xro = XRb[l];
    float* xvo = XVb[l];
    long nrt = (long)NR * 256, nvt = (long)NV * 256;
    k_init_bias<<<nblk(nrt, 256), 256, 0, stream>>>(xro, L[l][0].bias, L[l][1].bias, nrt, 256);
    k_init_bias<<<nblk(nvt, 256), 256, 0, stream>>>(xvo, L[l][2].bias, nullptr, nvt, 256);
    run_rel(stream, xrh, kr, NR, xrh, kr, NR, EI[0], E_[0], L[l][0], xro, BT, HS, HD, AS, AD, EMAX, DEN, PE);
    run_rel(stream, xvh, kv, NV, xrh, kr, NR, EI[1], E_[1], L[l][1], xro, BT, HS, HD, AS, AD, EMAX, DEN, PE);
    run_rel(stream, xrh, kr, NR, xvh, kv, NV, EI[2], E_[2], L[l][2], xvo, BT, HS, HD, AS, AD, EMAX, DEN, PE);
    k_combine<<<nblk(nrt, 256), 256, 0, stream>>>(xro, xr_res, 0.5f, nrt);
    k_combine<<<nblk(nvt, 256), 256, 0, stream>>>(xvo, xv_res, 1.0f, nvt);
    // BN + leaky(0.01), request then vehicle
    k_fill<<<2, 256, 0, stream>>>(SUM, 0.0f, 512);
    k_colstats<<<256, 256, 0, stream>>>(xro, NR, 256, SUM, SUM + 256);
    k_bn_final<<<1, 256, 0, stream>>>(SUM, SUM + 256, NR, 256, bng[l][0], bnb[l][0], nullptr, SCALE, SHIFT);
    k_bn_apply<<<nblk(nrt, 256), 256, 0, stream>>>(xro, SCALE, SHIFT, nrt, 256, 0.01f);
    k_fill<<<2, 256, 0, stream>>>(SUM, 0.0f, 512);
    k_colstats<<<256, 256, 0, stream>>>(xvo, NV, 256, SUM, SUM + 256);
    k_bn_final<<<1, 256, 0, stream>>>(SUM, SUM + 256, NV, 256, bng[l][1], bnb[l][1], nullptr, SCALE, SHIFT);
    k_bn_apply<<<nblk(nvt, 256), 256, 0, stream>>>(xvo, SCALE, SHIFT, nvt, 256, 0.01f);
    // f16 copies for next stage
    k_prep_h<<<nblk(nrt, 256), 256, 0, stream>>>(xro, XRH, nrt);
    k_prep_h<<<nblk(nvt, 256), 256, 0, stream>>>(xvo, XVH, nvt);
    xrh = XRH; kr = 256; xr_res = xro;
    xvh = XVH; kv = 256; xv_res = xvo;
  }

  // ---------------- edge head: out = [rr | vr | rv]
  float* outp = (float*)d_out;
  const _Float16* XSrc[3] = { XRH, XVH, XRH };
  const _Float16* XDst[3] = { XRH, XRH, XVH };
  const float* W1mid = lin1W + (long)256 * 128;
  // f16-transposed W1 blocks (x_s rows 0..255, x_d rows 384..639)
  k_prep_bt<<<nblk(128L * 256, 256), 256, 0, stream>>>(lin1W, WT1s, 256, 128, 128L * 256);
  k_prep_bt<<<nblk(128L * 256, 256), 256, 0, stream>>>(lin1W + (long)384 * 128, WT1d, 256, 128, 128L * 256);
  long off = 0;
  for (int t = 0; t < 3; ++t) {
    int E = E_[t];
    k_prep_h<<<nblk(16L * 128, 256), 256, 0, stream>>>(epW[t], EPWH, 16L * 128);
    gemm(stream, EPWH, W1mid, BT, CE, 16, 128, 128);        // Ce = Wp @ W1[256:384]
    k_prep_ct<<<nblk(128 * 32, 256), 256, 0, stream>>>(CE, CT);
    k_edge_dvec<<<1, 128, 0, stream>>>(epB[t], W1mid, lin1b, DV);
    k_prep_h<<<nblk((long)E * 16, 256), 256, 0, stream>>>(ea[t], EAH, (long)E * 16);
    k_fill<<<1, 256, 0, stream>>>(SUM, 0.0f, 256);
    long waves = (E + 15) / 16;
    k_edge_head<<<nblk(waves * 32, 256), 256, 0, stream>>>(
        EI[t], E, XSrc[t], XDst[t], EAH, WT1s, WT1d, CT, DV, 0, SUM, SUM + 128,
        nullptr, nullptr, lin2W, lin2b, nullptr);
    k_bn_final<<<1, 128, 0, stream>>>(SUM, SUM + 128, E, 128, fng, fnb, DV, SCALE, SHIFT);
    k_edge_head<<<nblk(waves * 32, 256), 256, 0, stream>>>(
        EI[t], E, XSrc[t], XDst[t], EAH, WT1s, WT1d, CT, DV, 1, nullptr, nullptr,
        SCALE, SHIFT, lin2W, lin2b, outp + off);
    off += E;
  }
}
